// MACUnit_33500744909512
// MI455X (gfx1250) — matrix operand order repaired and checked by reading, not compiled, not run on hardware
//
#include <hip/hip_runtime.h>
#include <hip/hip_bf16.h>
#include <math.h>

#define DD 512
#define BB 256
#define LL 30
#define KN 196
#define NSTEP 4
#define RR (BB * KN) // 50176
#define NKC 16       // 512 / 32 K-chunks

typedef __bf16 bf16_t;
typedef bf16_t v16bf __attribute__((ext_vector_type(16)));
typedef bf16_t v8bf  __attribute__((ext_vector_type(8)));
typedef float  v8f   __attribute__((ext_vector_type(8)));
typedef float  v4f   __attribute__((ext_vector_type(4)));

#define SP 72  // padded LDS row stride in bf16 (144B: 16B-aligned, conflict-free frag reads)

__device__ __forceinline__ v8f wmma_bf16(v16bf a, v16bf b, v8f c) {
    return __builtin_amdgcn_wmma_f32_16x16x32_bf16(false, a, false, b, (short)0, c, false, false);
}

// A fragment (16x32 bf16) from LDS per ISA layout:
// lane l: row m=l&15, half h=l>>4; elems 0..7 -> K=h*8+e ; elems 8..15 -> K=16+h*8+(e-8)
__device__ __forceinline__ v16bf load_a_frag_lds(const bf16_t* a_lds, int rowLocalBase, int lane) {
    int m = lane & 15;
    int h = lane >> 4;
    const bf16_t* p = a_lds + (rowLocalBase + m) * SP + h * 8;
    v8bf lo = *(const v8bf*)p;
    v8bf hi = *(const v8bf*)(p + 16);
    v16bf r;
#pragma unroll
    for (int i = 0; i < 8; ++i) { r[i] = lo[i]; r[i + 8] = hi[i]; }
    return r;
}

// B fragment (32x16 bf16) from transposed weights WT[N][K] (K contiguous):
// lane l: col n=l&15, half h=l>>4; elems 0..7 -> K = kc + h*8 + e ; elems 8..15 -> K = kc + 16 + h*8 + (e-8)
__device__ __forceinline__ v16bf load_b_frag(const bf16_t* __restrict__ WT, int nBase, int kc, int lane) {
    int n = nBase + (lane & 15);
    const bf16_t* p = WT + (size_t)n * DD + kc + (lane >> 4) * 8;
    v8bf lo = *(const v8bf*)p;
    v8bf hi = *(const v8bf*)(p + 16);
    v16bf r;
#pragma unroll
    for (int i = 0; i < 8; ++i) { r[i] = lo[i]; r[i + 8] = hi[i]; }
    return r;
}

// -------- weight transpose + bf16 convert: dst[n*512+k] = (bf16)src[(koff+k)*512+n]
__global__ void transpose_cvt_kernel(const float* __restrict__ src, int koff, bf16_t* __restrict__ dst) {
    int idx = blockIdx.x * 256 + threadIdx.x;
    int n = idx >> 9, k = idx & 511;
    dst[idx] = (bf16_t)src[(size_t)(koff + k) * DD + n];
}

__global__ void init_state_kernel(const float* __restrict__ ic, const float* __restrict__ im,
                                  float* __restrict__ control, float* __restrict__ memory) {
    int idx = blockIdx.x * 256 + threadIdx.x;
    int d = idx & (DD - 1);
    control[idx] = ic[d];
    memory[idx]  = im[d];
}

__global__ void logits_init_kernel(const float* __restrict__ rb, float* __restrict__ r_logits) {
    int idx = blockIdx.x * 256 + threadIdx.x;
    if (idx < RR) r_logits[idx] = rb[0];
}

// -------- GEMM0: know_proj = knowledge(f32) @ kprojT + bias -> bf16 [RR,512]
__global__ __launch_bounds__(256) void gemm_kproj_kernel(
    const float* __restrict__ A, const bf16_t* __restrict__ WT,
    const float* __restrict__ bias, bf16_t* __restrict__ out) {
    __shared__ bf16_t a_lds[2][64 * SP];
    int tid = threadIdx.x, lane = tid & 31, wid = tid >> 5;
    int wn = wid & 3, wm = wid >> 2;
    int rowBase = blockIdx.x * 64;
    int nBase = blockIdx.y * 256 + wn * 64;
    v8f acc[2][4] = {};
    int srow = tid >> 2, skp = (tid & 3) * 8;
    const float* gsrc = A + (size_t)(rowBase + srow) * DD + skp;
    int sidx = srow * SP + skp;

    // prologue: stage chunk 0
    v4f f0 = *(const v4f*)gsrc;
    v4f f1 = *(const v4f*)(gsrc + 4);
    {
        v8bf st;
#pragma unroll
        for (int i = 0; i < 4; ++i) { st[i] = (bf16_t)f0[i]; st[4 + i] = (bf16_t)f1[i]; }
        *(v8bf*)(&a_lds[0][sidx]) = st;
    }
    __syncthreads();

    for (int ic = 0; ic < NKC; ++ic) {
        int kc = ic * 32;
        int cur = ic & 1;
        if (ic < NKC - 1) {  // prefetch next chunk
            f0 = *(const v4f*)(gsrc + kc + 32);
            f1 = *(const v4f*)(gsrc + kc + 36);
        }
        v16bf af0 = load_a_frag_lds(a_lds[cur], wm * 32, lane);
        v16bf af1 = load_a_frag_lds(a_lds[cur], wm * 32 + 16, lane);
#pragma unroll
        for (int nt = 0; nt < 4; ++nt) {
            v16bf bf = load_b_frag(WT, nBase + nt * 16, kc, lane);
            acc[0][nt] = wmma_bf16(af0, bf, acc[0][nt]);
            acc[1][nt] = wmma_bf16(af1, bf, acc[1][nt]);
        }
        if (ic < NKC - 1) {
            v8bf st;
#pragma unroll
            for (int i = 0; i < 4; ++i) { st[i] = (bf16_t)f0[i]; st[4 + i] = (bf16_t)f1[i]; }
            *(v8bf*)(&a_lds[cur ^ 1][sidx]) = st;
            __syncthreads();
        }
    }
    int h = lane >> 4;
#pragma unroll
    for (int mt = 0; mt < 2; ++mt)
#pragma unroll
        for (int nt = 0; nt < 4; ++nt) {
            int n = nBase + nt * 16 + (lane & 15);
            float bv = bias[n];
#pragma unroll
            for (int v = 0; v < 8; ++v) {
                int row = rowBase + wm * 32 + mt * 16 + h * 8 + v;
                out[(size_t)row * DD + n] = (bf16_t)(acc[mt][nt][v] + bv);
            }
        }
}

// -------- GEMM1: ia1 = elu( (kp .* mp[b]) @ W1 + kp @ W2 + bias ) -> bf16
__global__ __launch_bounds__(256) void gemm_read1_kernel(
    const bf16_t* __restrict__ kp, const bf16_t* __restrict__ mp_bf,
    const bf16_t* __restrict__ W1T, const bf16_t* __restrict__ W2T,
    const float* __restrict__ bias, bf16_t* __restrict__ out) {
    __shared__ bf16_t a_lds[2][64 * SP];
    int tid = threadIdx.x, lane = tid & 31, wid = tid >> 5;
    int wn = wid & 3, wm = wid >> 2;
    int rowBase = blockIdx.x * 64;
    int nBase = blockIdx.y * 256 + wn * 64;
    int m = lane & 15, h = lane >> 4;
    v8f acc[2][4] = {};
    int srow = tid >> 2, skp = (tid & 3) * 8;
    const bf16_t* gsrc = kp + (size_t)(rowBase + srow) * DD + skp;
    int sidx = srow * SP + skp;

    // K-invariant per-lane mem_proj row pointers (batch index = row / 196)
    const bf16_t* mpbase[2];
#pragma unroll
    for (int mt = 0; mt < 2; ++mt) {
        int row = rowBase + wm * 32 + mt * 16 + m;
        mpbase[mt] = mp_bf + (size_t)(row / KN) * DD + h * 8;
    }

    v8bf streg = *(const v8bf*)gsrc;
    *(v8bf*)(&a_lds[0][sidx]) = streg;
    __syncthreads();

    for (int ic = 0; ic < NKC; ++ic) {
        int kc = ic * 32;
        int cur = ic & 1;
        if (ic < NKC - 1) streg = *(const v8bf*)(gsrc + kc + 32);
        v16bf kf[2], pf[2];
#pragma unroll
        for (int mt = 0; mt < 2; ++mt) {
            kf[mt] = load_a_frag_lds(a_lds[cur], wm * 32 + mt * 16, lane);
            const bf16_t* mpp = mpbase[mt] + kc;
            v8bf mlo = *(const v8bf*)mpp;
            v8bf mhi = *(const v8bf*)(mpp + 16);
            v16bf p;
#pragma unroll
            for (int i = 0; i < 8; ++i) {
                p[i]     = (bf16_t)((float)kf[mt][i]     * (float)mlo[i]);
                p[i + 8] = (bf16_t)((float)kf[mt][i + 8] * (float)mhi[i]);
            }
            pf[mt] = p;
        }
#pragma unroll
        for (int nt = 0; nt < 4; ++nt) {
            v16bf b1 = load_b_frag(W1T, nBase + nt * 16, kc, lane);
            v16bf b2 = load_b_frag(W2T, nBase + nt * 16, kc, lane);
#pragma unroll
            for (int mt = 0; mt < 2; ++mt) {
                acc[mt][nt] = wmma_bf16(pf[mt], b1, acc[mt][nt]);
                acc[mt][nt] = wmma_bf16(kf[mt], b2, acc[mt][nt]);
            }
        }
        if (ic < NKC - 1) {
            *(v8bf*)(&a_lds[cur ^ 1][sidx]) = streg;
            __syncthreads();
        }
    }
#pragma unroll
    for (int mt = 0; mt < 2; ++mt)
#pragma unroll
        for (int nt = 0; nt < 4; ++nt) {
            int n = nBase + nt * 16 + (lane & 15);
            float bv = bias[n];
#pragma unroll
            for (int v = 0; v < 8; ++v) {
                int row = rowBase + wm * 32 + mt * 16 + h * 8 + v;
                float z = acc[mt][nt][v] + bv;
                float e = z > 0.f ? z : expm1f(z);
                out[(size_t)row * DD + n] = (bf16_t)e;
            }
        }
}

// -------- GEMM2: z = (ia1 @ W3 + b3) * control[b]; e = elu(z); logits[row] += e . attn_w
__global__ __launch_bounds__(256) void gemm_read2_kernel(
    const bf16_t* __restrict__ ia1, const bf16_t* __restrict__ W3T,
    const float* __restrict__ b3, const float* __restrict__ control,
    const float* __restrict__ attn_w, float* __restrict__ r_logits) {
    __shared__ bf16_t a_lds[2][64 * SP];
    int tid = threadIdx.x, lane = tid & 31, wid = tid >> 5;
    int wn = wid & 3, wm = wid >> 2;
    int rowBase = blockIdx.x * 64;
    int nBase = blockIdx.y * 256 + wn * 64;
    int h = lane >> 4;
    v8f acc[2][4] = {};
    int srow = tid >> 2, skp = (tid & 3) * 8;
    const bf16_t* gsrc = ia1 + (size_t)(rowBase + srow) * DD + skp;
    int sidx = srow * SP + skp;

    v8bf streg = *(const v8bf*)gsrc;
    *(v8bf*)(&a_lds[0][sidx]) = streg;
    __syncthreads();

    for (int ic = 0; ic < NKC; ++ic) {
        int kc = ic * 32;
        int cur = ic & 1;
        if (ic < NKC - 1) streg = *(const v8bf*)(gsrc + kc + 32);
        v16bf af0 = load_a_frag_lds(a_lds[cur], wm * 32, lane);
        v16bf af1 = load_a_frag_lds(a_lds[cur], wm * 32 + 16, lane);
#pragma unroll
        for (int nt = 0; nt < 4; ++nt) {
            v16bf bf = load_b_frag(W3T, nBase + nt * 16, kc, lane);
            acc[0][nt] = wmma_bf16(af0, bf, acc[0][nt]);
            acc[1][nt] = wmma_bf16(af1, bf, acc[1][nt]);
        }
        if (ic < NKC - 1) {
            *(v8bf*)(&a_lds[cur ^ 1][sidx]) = streg;
            __syncthreads();
        }
    }
    float rsum[2][8];
#pragma unroll
    for (int mt = 0; mt < 2; ++mt)
#pragma unroll
        for (int v = 0; v < 8; ++v) rsum[mt][v] = 0.f;
#pragma unroll
    for (int mt = 0; mt < 2; ++mt)
#pragma unroll
        for (int nt = 0; nt < 4; ++nt) {
            int n = nBase + nt * 16 + (lane & 15);
            float bv = b3[n];
            float aw = attn_w[n];
#pragma unroll
            for (int v = 0; v < 8; ++v) {
                int row = rowBase + wm * 32 + mt * 16 + h * 8 + v;
                int bb = row / KN;
                float z = (acc[mt][nt][v] + bv) * control[(size_t)bb * DD + n];
                float e = z > 0.f ? z : expm1f(z);
                rsum[mt][v] += e * aw;
            }
        }
#pragma unroll
    for (int mt = 0; mt < 2; ++mt)
#pragma unroll
        for (int v = 0; v < 8; ++v) {
            float s = rsum[mt][v];
            s += __shfl_xor(s, 1);
            s += __shfl_xor(s, 2);
            s += __shfl_xor(s, 4);
            s += __shfl_xor(s, 8);
            if ((lane & 15) == 0) {
                int row = rowBase + wm * 32 + mt * 16 + h * 8 + v;
                atomicAdd(&r_logits[row], s);
            }
        }
}

// -------- small fp32 linear: out[b][d] = act( x1@W[0:512] (+ x2@W[512:1024]) + bias )
__global__ __launch_bounds__(256) void lin_kernel(
    const float* __restrict__ x1, const float* __restrict__ x2,
    const float* __restrict__ W, const float* __restrict__ bias,
    float* __restrict__ out, bf16_t* __restrict__ out_bf, int act) {
    int d = blockIdx.x * 256 + threadIdx.x;
    int b0 = blockIdx.y * 4;
    float a0, a1, a2, a3;
    a0 = a1 = a2 = a3 = bias[d];
#pragma unroll 4
    for (int j = 0; j < DD; ++j) {
        float w = W[(size_t)j * DD + d];
        a0 += x1[(size_t)(b0 + 0) * DD + j] * w;
        a1 += x1[(size_t)(b0 + 1) * DD + j] * w;
        a2 += x1[(size_t)(b0 + 2) * DD + j] * w;
        a3 += x1[(size_t)(b0 + 3) * DD + j] * w;
    }
    if (x2) {
#pragma unroll 4
        for (int j = 0; j < DD; ++j) {
            float w = W[(size_t)(DD + j) * DD + d];
            a0 += x2[(size_t)(b0 + 0) * DD + j] * w;
            a1 += x2[(size_t)(b0 + 1) * DD + j] * w;
            a2 += x2[(size_t)(b0 + 2) * DD + j] * w;
            a3 += x2[(size_t)(b0 + 3) * DD + j] * w;
        }
    }
    float r[4] = {a0, a1, a2, a3};
#pragma unroll
    for (int t = 0; t < 4; ++t) {
        float v = r[t];
        if (act == 1) v = tanhf(v);
        out[(size_t)(b0 + t) * DD + d] = v;
        if (out_bf) out_bf[(size_t)(b0 + t) * DD + d] = (bf16_t)v;
    }
}

// -------- control attention (block per batch)
__global__ __launch_bounds__(256) void ctrl_attn_kernel(
    const float* __restrict__ q2, const float* __restrict__ ctx,
    const float* __restrict__ caw, const float* __restrict__ cab,
    float* __restrict__ control) {
    int b = blockIdx.x, tid = threadIdx.x;
    int lane = tid & 31, wid = tid >> 5;
    __shared__ float attn[LL];
    for (int l = wid; l < LL; l += 8) {
        const float* cp = ctx + ((size_t)b * LL + l) * DD;
        float s = 0.f;
#pragma unroll
        for (int i = 0; i < 16; ++i) {
            int d = i * 32 + lane;
            s += q2[(size_t)b * DD + d] * cp[d] * caw[d];
        }
        s += __shfl_xor(s, 1); s += __shfl_xor(s, 2); s += __shfl_xor(s, 4);
        s += __shfl_xor(s, 8); s += __shfl_xor(s, 16);
        if (lane == 0) attn[l] = s + cab[0];
    }
    __syncthreads();
    if (tid == 0) {
        float mx = -1e30f;
        for (int l = 0; l < LL; ++l) mx = fmaxf(mx, attn[l]);
        float sm = 0.f;
        for (int l = 0; l < LL; ++l) { float e = __expf(attn[l] - mx); attn[l] = e; sm += e; }
        float inv = 1.0f / sm;
        for (int l = 0; l < LL; ++l) attn[l] *= inv;
    }
    __syncthreads();
    for (int d = tid; d < DD; d += 256) {
        float acc = 0.f;
        for (int l = 0; l < LL; ++l) acc += attn[l] * ctx[((size_t)b * LL + l) * DD + d];
        control[(size_t)b * DD + d] = acc;
    }
}

// -------- read softmax + weighted sum of knowledge (block per batch)
__global__ __launch_bounds__(256) void read_softmax_kernel(
    const float* __restrict__ r_logits, const float* __restrict__ knowledge,
    float* __restrict__ info) {
    int b = blockIdx.x, tid = threadIdx.x;
    __shared__ float sh[256];
    __shared__ float attn[KN];
    float v = (tid < KN) ? r_logits[(size_t)b * KN + tid] : -1e30f;
    sh[tid] = v; __syncthreads();
    for (int s = 128; s > 0; s >>= 1) { if (tid < s) sh[tid] = fmaxf(sh[tid], sh[tid + s]); __syncthreads(); }
    float mx = sh[0]; __syncthreads();
    float e = (tid < KN) ? __expf(v - mx) : 0.f;
    sh[tid] = e; __syncthreads();
    for (int s = 128; s > 0; s >>= 1) { if (tid < s) sh[tid] += sh[tid + s]; __syncthreads(); }
    float inv = 1.0f / sh[0];
    if (tid < KN) attn[tid] = e * inv;
    __syncthreads();
    for (int d = tid; d < DD; d += 256) {
        float acc = 0.f;
        for (int k = 0; k < KN; ++k) acc += attn[k] * knowledge[((size_t)b * KN + k) * DD + d];
        info[(size_t)b * DD + d] = acc;
    }
}

extern "C" void kernel_launch(void* const* d_in, const int* in_sizes, int n_in,
                              void* d_out, int out_size, void* d_ws, size_t ws_size,
                              hipStream_t stream) {
    const float* context     = (const float*)d_in[0];
    const float* knowledge   = (const float*)d_in[1];
    const float* ci_w        = (const float*)d_in[2];
    const float* ci_b        = (const float*)d_in[3];
    const float* cu_w        = (const float*)d_in[4];
    const float* cu_b        = (const float*)d_in[5];
    const float* ctrl_attn_w = (const float*)d_in[6];
    const float* ctrl_attn_b = (const float*)d_in[7];
    const float* kproj_w     = (const float*)d_in[8];
    const float* kproj_b     = (const float*)d_in[9];
    const float* mproj_w     = (const float*)d_in[10];
    const float* mproj_b     = (const float*)d_in[11];
    const float* rd_concat_w = (const float*)d_in[12];
    const float* rd_concat_b = (const float*)d_in[13];
    const float* rd_concat2_w= (const float*)d_in[14];
    const float* rd_concat2_b= (const float*)d_in[15];
    const float* rd_attn_w   = (const float*)d_in[16];
    const float* rd_attn_b   = (const float*)d_in[17];
    const float* wr_w        = (const float*)d_in[18];
    const float* wr_b        = (const float*)d_in[19];
    const float* init_ctrl   = (const float*)d_in[20];
    const float* init_mem    = (const float*)d_in[21];

    char* ws = (char*)d_ws;
    size_t off = 0;
    auto alloc = [&](size_t bytes) -> void* {
        void* p = (void*)(ws + off);
        off += (bytes + 255) & ~(size_t)255;
        return p;
    };
    bf16_t* kp     = (bf16_t*)alloc((size_t)RR * DD * 2);
    bf16_t* ia1    = (bf16_t*)alloc((size_t)RR * DD * 2);
    bf16_t* kprojT = (bf16_t*)alloc((size_t)DD * DD * 2);
    bf16_t* W1T    = (bf16_t*)alloc((size_t)DD * DD * 2);
    bf16_t* W2T    = (bf16_t*)alloc((size_t)DD * DD * 2);
    bf16_t* W3T    = (bf16_t*)alloc((size_t)DD * DD * 2);
    bf16_t* mp_bf  = (bf16_t*)alloc((size_t)BB * DD * 2);
    float*  q      = (float*)alloc((size_t)BB * DD * 4);
    float*  q2     = (float*)alloc((size_t)BB * DD * 4);
    float*  mp     = (float*)alloc((size_t)BB * DD * 4);
    float*  control= (float*)alloc((size_t)BB * DD * 4);
    float*  mem0   = (float*)alloc((size_t)BB * DD * 4);
    float*  mem1   = (float*)alloc((size_t)BB * DD * 4);
    float*  info   = (float*)alloc((size_t)BB * DD * 4);
    float*  r_logits = (float*)alloc((size_t)RR * 4);

    // bf16 transposed weights
    transpose_cvt_kernel<<<DD * DD / 256, 256, 0, stream>>>(kproj_w, 0, kprojT);
    transpose_cvt_kernel<<<DD * DD / 256, 256, 0, stream>>>(rd_concat_w, 0, W1T);
    transpose_cvt_kernel<<<DD * DD / 256, 256, 0, stream>>>(rd_concat_w, DD, W2T);
    transpose_cvt_kernel<<<DD * DD / 256, 256, 0, stream>>>(rd_concat2_w, 0, W3T);
    init_state_kernel<<<BB * DD / 256, 256, 0, stream>>>(init_ctrl, init_mem, control, mem0);

    // hoisted: know_proj (bf16)
    gemm_kproj_kernel<<<dim3(RR / 64, 2), 256, 0, stream>>>(knowledge, kprojT, kproj_b, kp);

    float* mcur = mem0;
    float* mnext = mem1;
    for (int step = 0; step < NSTEP; ++step) {
        // ControlUnit
        lin_kernel<<<dim3(2, BB / 4), 256, 0, stream>>>(control, mcur, ci_w, ci_b, q, nullptr, 1);
        lin_kernel<<<dim3(2, BB / 4), 256, 0, stream>>>(q, nullptr, cu_w + (size_t)step * DD * DD,
                                                        cu_b + (size_t)step * DD, q2, nullptr, 0);
        ctrl_attn_kernel<<<BB, 256, 0, stream>>>(q2, context, ctrl_attn_w, ctrl_attn_b, control);
        // ReadUnit
        lin_kernel<<<dim3(2, BB / 4), 256, 0, stream>>>(mcur, nullptr, mproj_w, mproj_b, mp, mp_bf, 0);
        gemm_read1_kernel<<<dim3(RR / 64, 2), 256, 0, stream>>>(kp, mp_bf, W1T, W2T, rd_concat_b, ia1);
        logits_init_kernel<<<(RR + 255) / 256, 256, 0, stream>>>(rd_attn_b, r_logits);
        gemm_read2_kernel<<<dim3(RR / 64, 2), 256, 0, stream>>>(ia1, W3T, rd_concat2_b, control,
                                                                rd_attn_w, r_logits);
        read_softmax_kernel<<<BB, 256, 0, stream>>>(r_logits, knowledge, info);
        // WriteUnit
        lin_kernel<<<dim3(2, BB / 4), 256, 0, stream>>>(mcur, info, wr_w, wr_b, mnext, nullptr, 0);
        float* t = mcur; mcur = mnext; mnext = t;
    }
    hipMemcpyAsync(d_out, mcur, (size_t)BB * DD * sizeof(float),
                   hipMemcpyDeviceToDevice, stream);
}
